// MPNEncoder_25924422599323
// MI455X (gfx1250) — compile-verified, matrix operand order checked by reading, not run on hardware
//
#include <hip/hip_runtime.h>
#include <hip/hip_bf16.h>

typedef __attribute__((ext_vector_type(16))) __bf16 v16bf;
typedef __attribute__((ext_vector_type(4)))  __bf16 v4bf;
typedef __attribute__((ext_vector_type(4)))  float  v4f;
typedef __attribute__((ext_vector_type(8)))  float  v8f;

#define NATOMS 40000
#define NBONDS 90000
#define MAXNB  6
#define AFD    133
#define BFD    147
#define HID    300
#define DEPTH  6

// padded sizes (K to mult of 32, M to mult of 128, N to mult of 64)
#define KB_PAD 160   // BFD 147 -> 160
#define KH_PAD 320   // HID 300 -> 320
#define KO_PAD 448   // AFD+HID 433 -> 448
#define MB_PAD 90112 // NBONDS -> mult of 128
#define MA_PAD 40064 // NATOMS -> mult of 128
#define NH_PAD 320   // N=300 -> 320 (weight row pad)
#define N3_PAD 960   // N=900 -> 960

#define LDP 36       // LDS tile pitch (bf16): 72B rows -> 8B-aligned b64 stores,
                     // 18-bank stride -> conflict-free fragment reads
#define A_TILE (128 * LDP)
#define B_TILE (64 * LDP)

// ---------------------------------------------------------------------------
__device__ __forceinline__ v8f wmma_bf16(v16bf a, v16bf b, v8f c) {
  return __builtin_amdgcn_wmma_f32_16x16x32_bf16(false, a, false, b, (short)0, c,
                                                 false, false);
}

// 16x32 bf16 fragment from LDS tile (row pitch LDP). A layout per ISA 7.12.2;
// B uses the same per-lane pattern with N in place of M.
__device__ __forceinline__ v16bf load_frag(const __bf16* rowp) {
  const int lane = threadIdx.x & 31;
  const int half = lane >> 4;
  const int m    = lane & 15;
  const __bf16* rp = rowp + m * LDP;
  v16bf f;
#pragma unroll
  for (int v = 0; v < 8; ++v) {
    int k = ((v >> 2) << 4) + half * 8 + ((v & 3) << 1);
    f[2 * v]     = rp[k];
    f[2 * v + 1] = rp[k + 1];
  }
  return f;
}

__device__ __forceinline__ uint2 pk_bf16x4(float4 f) {
  v4f x; x[0] = f.x; x[1] = f.y; x[2] = f.z; x[3] = f.w;
  v4bf h = __builtin_convertvector(x, v4bf);   // -> 2x v_cvt_pk_bf16_f32
  union { v4bf h; uint2 u; } u; u.h = h;
  return u.u;
}

// ---------------------------------------------------------------------------
// C[M,N](f32) = act(A[Mpad,lda](bf16) @ W[Npad,ldw](bf16)^T + bias) * rowscale
// Optionally also stores bf16 copy of C into Cbf (padded, pre-zeroed).
// Both operands pre-converted/padded: fills are pure uint4 copies, no VALU.
// Tile 128x64, BK=32, LDS double-buffered, one barrier per K-step.
// 8 waves: wm=0..3 x wn=0..1, each 2x2 WMMA tiles.
// ---------------------------------------------------------------------------
__global__ void gemm_bf16_kernel(const __bf16* __restrict__ A, int lda,
                                 const __bf16* __restrict__ W, int ldw,
                                 const float* __restrict__ bias,
                                 const float* __restrict__ rowscale,
                                 float* __restrict__ C, int ldc,
                                 __bf16* __restrict__ Cbf, int ldcb,
                                 int M, int N, int Kpad, int act) {
  __shared__ __bf16 As[2 * A_TILE];
  __shared__ __bf16 Bs[2 * B_TILE];

  const int m0   = blockIdx.y * 128;
  const int n0   = blockIdx.x * 64;
  const int tid  = threadIdx.x;
  const int wave = tid >> 5;
  const int lane = tid & 31;
  const int wm   = wave >> 1;   // 0..3
  const int wn   = wave & 1;    // 0..1

  // per-thread fill pointers (loop-invariant). uint4 = 8 bf16.
  const __bf16* Ap[2];
  __bf16*       Asl[2];
#pragma unroll
  for (int p = 0; p < 2; ++p) {
    int vec = tid + p * 256;          // 0..511
    int r   = vec >> 2;               // 0..127 (4 uint4 per 32-col row)
    int c8  = (vec & 3) << 3;         // 0,8,16,24
    Ap[p]  = A + (size_t)(m0 + r) * lda + c8;
    Asl[p] = &As[r * LDP + c8];
  }
  const int br  = tid >> 2;           // 0..63
  const int bc8 = (tid & 3) << 3;
  const __bf16* Wp  = W + (size_t)(n0 + br) * ldw + bc8;
  __bf16*       Bsl = &Bs[br * LDP + bc8];

  v8f acc[4];
#pragma unroll
  for (int t = 0; t < 4; ++t)
#pragma unroll
    for (int i = 0; i < 8; ++i) acc[t][i] = 0.f;

  const int nsteps = Kpad >> 5;
  uint4 areg[2], breg;

  auto sstore = [&](int buf) {
#pragma unroll
    for (int p = 0; p < 2; ++p) {
      __bf16* d = Asl[p] + buf * A_TILE;
      *(uint2*)(d)     = make_uint2(areg[p].x, areg[p].y);
      *(uint2*)(d + 4) = make_uint2(areg[p].z, areg[p].w);
    }
    __bf16* d = Bsl + buf * B_TILE;
    *(uint2*)(d)     = make_uint2(breg.x, breg.y);
    *(uint2*)(d + 4) = make_uint2(breg.z, breg.w);
  };

  // prologue: tile 0 -> buffer 0
  areg[0] = *(const uint4*)(Ap[0]);
  areg[1] = *(const uint4*)(Ap[1]);
  breg    = *(const uint4*)(Wp);
  sstore(0);
  __syncthreads();

  for (int s = 0; s < nsteps; ++s) {
    const int  cur  = s & 1;
    const bool more = (s + 1) < nsteps;
    const int  ktn  = (s + 1) << 5;

    if (more) {
      areg[0] = *(const uint4*)(Ap[0] + ktn);
      areg[1] = *(const uint4*)(Ap[1] + ktn);
      breg    = *(const uint4*)(Wp + ktn);
      if (s + 2 < nsteps)  // gfx1250 global_prefetch_b8 on A stream
        __builtin_prefetch(Ap[0] + ((s + 2) << 5), 0, 1);
    }

    const __bf16* Ab = &As[cur * A_TILE];
    const __bf16* Bb = &Bs[cur * B_TILE];
    v16bf a0 = load_frag(Ab + (wm * 32) * LDP);
    v16bf a1 = load_frag(Ab + (wm * 32 + 16) * LDP);
    v16bf b0 = load_frag(Bb + (wn * 32) * LDP);
    v16bf b1 = load_frag(Bb + (wn * 32 + 16) * LDP);
    acc[0] = wmma_bf16(a0, b0, acc[0]);
    acc[1] = wmma_bf16(a0, b1, acc[1]);
    acc[2] = wmma_bf16(a1, b0, acc[2]);
    acc[3] = wmma_bf16(a1, b1, acc[3]);

    if (more) sstore(cur ^ 1);
    __syncthreads();
  }

  const int half = lane >> 4;
  const int nl   = lane & 15;
#pragma unroll
  for (int tm = 0; tm < 2; ++tm) {
#pragma unroll
    for (int tn = 0; tn < 2; ++tn) {
      const v8f& a = acc[tm * 2 + tn];
      int nn = n0 + wn * 32 + tn * 16 + nl;
      if (nn >= N) continue;
      float bv = bias ? bias[nn] : 0.f;
#pragma unroll
      for (int v = 0; v < 8; ++v) {
        int mm = m0 + wm * 32 + tm * 16 + v + half * 8;
        if (mm >= M) continue;
        float val = a[v] + bv;
        if (act == 1) val = fmaxf(val, 0.f);
        float rs = rowscale ? rowscale[mm] : 1.f;
        C[(size_t)mm * ldc + nn] = val * rs;
        if (Cbf) Cbf[(size_t)mm * ldcb + nn] = (__bf16)(val * rs);
      }
    }
  }
}

// ---------------------------------------------------------------------------
// Helper kernels
// ---------------------------------------------------------------------------
__global__ void fill_zero_u32(unsigned int* __restrict__ p, long long n) {
  long long i = (long long)blockIdx.x * blockDim.x + threadIdx.x;
  if (i < n) p[i] = 0u;
}

__global__ void cvt_pad_bf16_kernel(const float* __restrict__ src,
                                    __bf16* __restrict__ dst,
                                    int N, int K, int Kpad) {
  int t = blockIdx.x * blockDim.x + threadIdx.x;
  if (t >= N * K) return;
  int n = t / K, k = t % K;
  dst[(size_t)n * Kpad + k] = (__bf16)src[t];
}

__global__ void gather_sum_kernel(const float* __restrict__ msg,
                                  const int* __restrict__ a2b,
                                  float* __restrict__ amsg) {
  const int C4 = HID / 4;  // 75
  int t = blockIdx.x * blockDim.x + threadIdx.x;
  if (t >= NATOMS * C4) return;
  int a = t / C4, c = t % C4;
  float4 s = make_float4(0.f, 0.f, 0.f, 0.f);
#pragma unroll
  for (int k = 0; k < MAXNB; ++k) {
    int b = a2b[a * MAXNB + k];
    float4 v = ((const float4*)(msg + (size_t)b * HID))[c];
    s.x += v.x; s.y += v.y; s.z += v.z; s.w += v.w;
  }
  ((float4*)(amsg + (size_t)a * HID))[c] = s;
}

// h = a_message[b2a] - message[b2revb]; writes f32 (for GRU) + padded bf16 (GEMM A)
__global__ void bond_update_kernel(const float* __restrict__ amsg,
                                   const float* __restrict__ msg,
                                   const int* __restrict__ b2a,
                                   const int* __restrict__ b2revb,
                                   float* __restrict__ hout /* ld=HID */,
                                   __bf16* __restrict__ hbf /* ld=KH_PAD */) {
  const int C4 = HID / 4;
  int t = blockIdx.x * blockDim.x + threadIdx.x;
  if (t >= NBONDS * C4) return;
  int b = t / C4, c = t % C4;
  int a = b2a[b], rb = b2revb[b];
  float4 x = ((const float4*)(amsg + (size_t)a * HID))[c];
  float4 y = ((const float4*)(msg + (size_t)rb * HID))[c];
  float4 o = make_float4(x.x - y.x, x.y - y.y, x.z - y.z, x.w - y.w);
  ((float4*)(hout + (size_t)b * HID))[c] = o;
  *(uint2*)(hbf + (size_t)b * KH_PAD + c * 4) = pk_bf16x4(o);
}

__global__ void gru_kernel(const float* __restrict__ gi,
                           const float* __restrict__ gh,
                           const float* __restrict__ h /* ld=HID */,
                           float* __restrict__ out /* ld=HID */) {
  int t = blockIdx.x * blockDim.x + threadIdx.x;
  if (t >= NBONDS * HID) return;
  int row = t / HID, c = t % HID;
  size_t base = (size_t)row * 3 * HID;
  float ir = gi[base + c], iz = gi[base + HID + c], inn = gi[base + 2 * HID + c];
  float hr = gh[base + c], hz = gh[base + HID + c], hn  = gh[base + 2 * HID + c];
  float r = 1.f / (1.f + __expf(-(ir + hr)));
  float z = 1.f / (1.f + __expf(-(iz + hz)));
  float n = tanhf(inn + r * hn);
  float o = (1.f - z) * n + z * h[(size_t)row * HID + c];
  out[(size_t)row * HID + c] = (row == 0) ? 0.f : o;  // msg_mask
}

// a_input = [f_atoms | a_message] -> bf16 padded (GEMM A)
__global__ void concat_kernel(const float* __restrict__ fa,
                              const float* __restrict__ amsg,
                              __bf16* __restrict__ ain /* ld=KO_PAD */) {
  const int W = AFD + HID;  // 433
  int t = blockIdx.x * blockDim.x + threadIdx.x;
  if (t >= NATOMS * W) return;
  int a = t / W, c = t % W;
  float v = (c < AFD) ? fa[(size_t)a * AFD + c] : amsg[(size_t)a * HID + (c - AFD)];
  ain[(size_t)a * KO_PAD + c] = (__bf16)v;
}

// ---------------------------------------------------------------------------
extern "C" void kernel_launch(void* const* d_in, const int* in_sizes, int n_in,
                              void* d_out, int out_size, void* d_ws, size_t ws_size,
                              hipStream_t stream) {
  const float* f_atoms = (const float*)d_in[0];
  const float* f_bonds = (const float*)d_in[1];
  const int*   a2b     = (const int*)d_in[2];
  const int*   b2a     = (const int*)d_in[3];
  const int*   b2revb  = (const int*)d_in[4];
  const float* mask    = (const float*)d_in[8];
  const float* W_i     = (const float*)d_in[9];
  const float* W_ih    = (const float*)d_in[10];
  const float* W_hh    = (const float*)d_in[11];
  const float* b_ih    = (const float*)d_in[12];
  const float* b_hh    = (const float*)d_in[13];
  const float* W_o_w   = (const float*)d_in[14];
  const float* W_o_b   = (const float*)d_in[15];
  float* out = (float*)d_out;

  char* ws = (char*)d_ws;
  auto carve = [&](size_t bytes) {
    char* p = ws;
    ws += (bytes + 255) & ~(size_t)255;
    return (void*)p;
  };
  __bf16* Wi_bf  = (__bf16*)carve((size_t)NH_PAD * KB_PAD * 2);
  __bf16* Wih_bf = (__bf16*)carve((size_t)N3_PAD * KH_PAD * 2);
  __bf16* Whh_bf = (__bf16*)carve((size_t)N3_PAD * KH_PAD * 2);
  __bf16* Wo_bf  = (__bf16*)carve((size_t)NH_PAD * KO_PAD * 2);
  __bf16* fb_bf  = (__bf16*)carve((size_t)MB_PAD * KB_PAD * 2);
  __bf16* inp_bf = (__bf16*)carve((size_t)MB_PAD * KH_PAD * 2);
  __bf16* hb_bf  = (__bf16*)carve((size_t)MB_PAD * KH_PAD * 2);
  __bf16* ain_bf = (__bf16*)carve((size_t)MA_PAD * KO_PAD * 2);
  float* message = (float*)carve((size_t)NBONDS * HID * 4);
  float* hbuf    = (float*)carve((size_t)NBONDS * HID * 4);
  float* gi      = (float*)carve((size_t)NBONDS * 3 * HID * 4);
  float* gh      = (float*)carve((size_t)NBONDS * 3 * HID * 4);
  float* amsg    = (float*)carve((size_t)NATOMS * HID * 4);

  auto zero = [&](void* p, long long n_u32) {
    fill_zero_u32<<<(unsigned)((n_u32 + 255) / 256), 256, 0, stream>>>(
        (unsigned int*)p, n_u32);
  };
  // padding must read as 0 in the unguarded GEMM loads (bf16 -> /2 u32 counts)
  zero(Wi_bf,  (long long)NH_PAD * KB_PAD / 2);
  zero(Wih_bf, (long long)N3_PAD * KH_PAD / 2);
  zero(Whh_bf, (long long)N3_PAD * KH_PAD / 2);
  zero(Wo_bf,  (long long)NH_PAD * KO_PAD / 2);
  zero(fb_bf,  (long long)MB_PAD * KB_PAD / 2);
  zero(inp_bf, (long long)MB_PAD * KH_PAD / 2);
  zero(hb_bf,  (long long)MB_PAD * KH_PAD / 2);
  zero(ain_bf, (long long)MA_PAD * KO_PAD / 2);

  auto cvt = [&](const float* s, __bf16* d, int N, int K, int Kpad) {
    cvt_pad_bf16_kernel<<<(N * K + 255) / 256, 256, 0, stream>>>(s, d, N, K, Kpad);
  };
  cvt(W_i,     Wi_bf,  HID,     BFD,       KB_PAD);
  cvt(W_ih,    Wih_bf, 3 * HID, HID,       KH_PAD);
  cvt(W_hh,    Whh_bf, 3 * HID, HID,       KH_PAD);
  cvt(W_o_w,   Wo_bf,  HID,     AFD + HID, KO_PAD);
  cvt(f_bonds, fb_bf,  NBONDS,  BFD,       KB_PAD);  // A of first GEMM

  auto gemm = [&](const __bf16* A, int lda, const __bf16* W, int ldw,
                  const float* bias, const float* rs, float* C, int ldc,
                  __bf16* Cbf, int ldcb, int M, int N, int Kpad, int act) {
    dim3 grid((N + 63) / 64, (M + 127) / 128);
    gemm_bf16_kernel<<<grid, dim3(256), 0, stream>>>(A, lda, W, ldw, bias, rs,
                                                     C, ldc, Cbf, ldcb,
                                                     M, N, Kpad, act);
  };

  // message = inp = f_bonds @ W_i^T (f32 + bf16 copies in one pass)
  gemm(fb_bf, KB_PAD, Wi_bf, KB_PAD, nullptr, nullptr, message, HID,
       inp_bf, KH_PAD, NBONDS, HID, KB_PAD, 0);
  // gi = inp @ W_ih^T + b_ih (loop invariant, hoisted)
  gemm(inp_bf, KH_PAD, Wih_bf, KH_PAD, b_ih, nullptr, gi, 3 * HID,
       nullptr, 0, NBONDS, 3 * HID, KH_PAD, 0);

  const int Ta = NATOMS * (HID / 4);
  const int Tb = NBONDS * (HID / 4);
  for (int it = 0; it < DEPTH - 1; ++it) {
    gather_sum_kernel<<<(Ta + 255) / 256, 256, 0, stream>>>(message, a2b, amsg);
    bond_update_kernel<<<(Tb + 255) / 256, 256, 0, stream>>>(amsg, message, b2a,
                                                             b2revb, hbuf, hb_bf);
    gemm(hb_bf, KH_PAD, Whh_bf, KH_PAD, b_hh, nullptr, gh, 3 * HID,
         nullptr, 0, NBONDS, 3 * HID, KH_PAD, 0);
    gru_kernel<<<(NBONDS * HID + 255) / 256, 256, 0, stream>>>(gi, gh, hbuf,
                                                               message);
  }

  gather_sum_kernel<<<(Ta + 255) / 256, 256, 0, stream>>>(message, a2b, amsg);
  concat_kernel<<<(NATOMS * (AFD + HID) + 255) / 256, 256, 0, stream>>>(
      f_atoms, amsg, ain_bf);
  // out = relu(ain @ W_o^T + b) * mask
  gemm(ain_bf, KO_PAD, Wo_bf, KO_PAD, W_o_b, mask, out, HID,
       nullptr, 0, NATOMS, HID, KO_PAD, 1);
}
